// BertClassifier_77309411685
// MI455X (gfx1250) — matrix operand order repaired and checked by reading, not compiled, not run on hardware
//
#include <hip/hip_runtime.h>
#include <hip/hip_bf16.h>

typedef __attribute__((ext_vector_type(16))) _Float16 v16h;
typedef __attribute__((ext_vector_type(8)))  _Float16 v8h;
typedef __attribute__((ext_vector_type(8)))  float    v8f;

#define SEQ 256
#define H 768
#define K3H 2304        // (NB_CTX+1)*H
#define INNER 256
#define NB_LABELS 3
// Static context positions: np.random.default_rng(0).choice(256, size=2)
// (PCG64 seed 0, Lemire bounded sampling -> high 8 bits of first two uint64s)
#define CTX0 163
#define CTX1 69

// ---------------------------------------------------------------------------
// Kernel 1: feat[b, :] = concat(mean(emb[b, start:stop, :]), emb[b,CTX0,:], emb[b,CTX1,:])
// written directly in f16 to halve GEMM-side read traffic.
// ---------------------------------------------------------------------------
__global__ void build_feat(const float* __restrict__ emb,
                           const int*   __restrict__ pos,
                           _Float16*    __restrict__ feat) {
  const int b = blockIdx.x;
  const int t = threadIdx.x;                      // 256 threads
  const int start = pos[2 * b];
  const int stop  = pos[2 * b + 1];
  const float inv = 1.0f / (float)(stop - start); // span length 1..8
  const float* eb = emb + (size_t)b * SEQ * H;
  _Float16* fb = feat + (size_t)b * K3H;
  for (int c = t; c < H; c += 256) {
    float s = 0.0f;
    for (int r = start; r < stop; ++r) s += eb[(size_t)r * H + c];
    fb[c]         = (_Float16)(s * inv);
    fb[H + c]     = (_Float16)eb[(size_t)CTX0 * H + c];
    fb[2 * H + c] = (_Float16)eb[(size_t)CTX1 * H + c];
  }
}

// ---------------------------------------------------------------------------
// Kernel 2: one-time f32 -> f16 conversion of W_base [INNER, K3H] (row-major,
// row n holds the K-dim contiguously -> perfect for the B-matrix lane loads).
// ---------------------------------------------------------------------------
__global__ void conv_w(const float* __restrict__ w, _Float16* __restrict__ w16, int n) {
  int i = blockIdx.x * 256 + threadIdx.x;
  if (i < n) w16[i] = (_Float16)w[i];
}

// ---------------------------------------------------------------------------
// Kernel 3: hidden = relu(feat @ W_base^T + b_base) via v_wmma_f32_16x16x32_f16.
// Block = 128 threads = 4 waves; wave w computes one 16x16 tile of the output:
//   rows   mBase..mBase+15  (mBase = blockIdx.x*16)
//   cols   nBase..nBase+15  (nBase = blockIdx.y*64 + w*16)
// K-loop: 2304/32 = 72 WMMA ops per wave, f32 accumulation.
//
// Per-lane layouts (CDNA5 ISA 7.12.2, wave32):
//   A (16x32 f16): lane L, row = L%16; halves 0..7  -> K = k0 + 8*(L>>4)   + j
//                                      halves 8..15 -> K = k0 + 16 + 8*(L>>4) + j
//   B (32x16 f16): lane L, col = L%16; same K mapping as A (8*(L>>4)+j, 16+8*(L>>4)+j)
//       B[k][n] = W_base[n][k]  (row-major W gives two contiguous 16B runs per lane)
//   C/D (16x16 f32): lane L, col = L%16; acc[i] -> row = i + 8*(L>>4)
// ---------------------------------------------------------------------------
__global__ void gemm_wmma(const _Float16* __restrict__ feat,
                          const _Float16* __restrict__ w16,
                          const float*    __restrict__ bias,
                          float*          __restrict__ hidden) {
  const int lane = threadIdx.x & 31;
  const int wave = threadIdx.x >> 5;
  const int grp  = lane >> 4;      // 0: lanes 0-15, 1: lanes 16-31
  const int ln   = lane & 15;
  const int mBase = blockIdx.x * 16;
  const int nBase = blockIdx.y * 64 + wave * 16;

  const _Float16* aRow = feat + (size_t)(mBase + ln) * K3H;  // A row for this lane
  const _Float16* bRow = w16  + (size_t)(nBase + ln) * K3H;  // B column (= W row)

  v8f acc = {};
  for (int k0 = 0; k0 < K3H; k0 += 32) {
    v8h a0 = *(const v8h*)(aRow + k0 + 8 * grp);        // 16B load
    v8h a1 = *(const v8h*)(aRow + k0 + 16 + 8 * grp);   // 16B load
    v16h a = __builtin_shufflevector(a0, a1,
        0, 1, 2, 3, 4, 5, 6, 7, 8, 9, 10, 11, 12, 13, 14, 15);
    v8h b0 = *(const v8h*)(bRow + k0 + 8 * grp);        // 16B load
    v8h b1 = *(const v8h*)(bRow + k0 + 16 + 8 * grp);   // 16B load
    v16h bm = __builtin_shufflevector(b0, b1,
        0, 1, 2, 3, 4, 5, 6, 7, 8, 9, 10, 11, 12, 13, 14, 15);
    acc = __builtin_amdgcn_wmma_f32_16x16x32_f16(
        /*neg_a=*/false, a, /*neg_b=*/false, bm,
        /*c_mod=*/(short)0, acc, /*reuse_a=*/false, /*reuse_b=*/false);
  }

  const float bv = bias[nBase + ln];
  #pragma unroll
  for (int i = 0; i < 8; ++i) {
    const int row = mBase + i + 8 * grp;
    float v = acc[i] + bv;
    hidden[(size_t)row * INNER + (nBase + ln)] = v > 0.0f ? v : 0.0f;
  }
}

// ---------------------------------------------------------------------------
// Kernel 4: per-sample expert head. out[b,n] = <hidden[b,:], W_e[cat[b],n,:]> + b_e.
// One block (256 threads) per sample; LDS tree reduction over 3 labels.
// ---------------------------------------------------------------------------
__global__ void expert_head(const float* __restrict__ hidden,
                            const int*   __restrict__ cat,
                            const float* __restrict__ We,
                            const float* __restrict__ be,
                            float*       __restrict__ out) {
  const int b = blockIdx.x;
  const int t = threadIdx.x;                 // 256 = INNER
  const int c = cat[b];
  const float hv = hidden[(size_t)b * INNER + t];
  const float* w = We + (size_t)c * NB_LABELS * INNER;
  __shared__ float red[256];
  for (int n = 0; n < NB_LABELS; ++n) {
    red[t] = hv * w[(size_t)n * INNER + t];
    __syncthreads();
    for (int off = 128; off > 0; off >>= 1) {
      if (t < off) red[t] += red[t + off];
      __syncthreads();
    }
    if (t == 0) out[(size_t)b * NB_LABELS + n] = red[0] + be[c * NB_LABELS + n];
    __syncthreads();
  }
}

// ---------------------------------------------------------------------------
extern "C" void kernel_launch(void* const* d_in, const int* in_sizes, int n_in,
                              void* d_out, int out_size, void* d_ws, size_t ws_size,
                              hipStream_t stream) {
  const float* emb = (const float*)d_in[0];
  const int*   pos = (const int*)  d_in[1];
  const int*   cat = (const int*)  d_in[2];
  const float* Wb  = (const float*)d_in[3];
  const float* bb  = (const float*)d_in[4];
  const float* We  = (const float*)d_in[5];
  const float* be  = (const float*)d_in[6];
  float* out = (float*)d_out;

  const int B = in_sizes[2];  // categories has one entry per sample

  // Workspace layout: feat f16 [B,2304] | W16 f16 [256,2304] | hidden f32 [B,256]
  char* ws = (char*)d_ws;
  _Float16* feat  = (_Float16*)ws;
  _Float16* w16   = (_Float16*)(ws + (size_t)B * K3H * sizeof(_Float16));
  float*    hidden = (float*)  (ws + (size_t)B * K3H * sizeof(_Float16)
                                   + (size_t)INNER * K3H * sizeof(_Float16));

  build_feat<<<B, 256, 0, stream>>>(emb, pos, feat);
  conv_w<<<(INNER * K3H + 255) / 256, 256, 0, stream>>>(Wb, w16, INNER * K3H);
  gemm_wmma<<<dim3(B / 16, INNER / 64), 128, 0, stream>>>(feat, w16, bb, hidden);
  expert_head<<<B, 256, 0, stream>>>(hidden, cat, We, be, out);
}
